// UResidualMamba_68564857913683
// MI455X (gfx1250) — matrix operand order repaired and checked by reading, not compiled, not run on hardware
//
#include <hip/hip_runtime.h>

// ---------------------------------------------------------------------------
// UResidualMamba forward for gfx1250 (MI455X, wave32, WMMA + TDM).
// GEMMs: operands pre-packed to f16 WMMA-fragment order. Per block, the shared
// B tile chain is DMA'd into LDS by the Tensor Data Mover (tensor_load_to_lds
// + s_wait_tensorcnt); the hot loop is global_load_b128 (A) + ds_load_b128 (B)
// + v_wmma_f32_16x16x32_f16 (f32 accumulate).
// Selective scan: register-resident 16-state recurrence per (batch,channel)
// thread with global_prefetch_b8 hints. Activations token-major (B*L, C).
// ---------------------------------------------------------------------------

typedef __attribute__((ext_vector_type(16))) _Float16     v16h;
typedef __attribute__((ext_vector_type(8)))  float        v8f;
typedef __attribute__((ext_vector_type(4)))  unsigned int v4u;
typedef __attribute__((ext_vector_type(8)))  int          v8i_;
typedef __attribute__((ext_vector_type(4)))  int          v4i_;

#define BATCH_C 4
#define SEQ_C   2048
#define TOK_C   (BATCH_C * SEQ_C)   // 8192 tokens (multiple of 16)
#define TT_C    (TOK_C / 16)        // 512 token tiles
#define NST     16
#define DCONV   4
#define LD_DBC  48                  // fixed (padded) ld for dbc buffer
#define MAX_KT  16                  // max K tiles (K <= 512)

#if defined(__HIP_DEVICE_COMPILE__) && __has_builtin(__builtin_amdgcn_tensor_load_to_lds)
#define USE_TDM 1
#else
#define USE_TDM 0
#endif

__device__ __forceinline__ float silu_f(float v)     { return v / (1.f + __expf(-v)); }
__device__ __forceinline__ float softplus_f(float v) { return (v > 20.f) ? v : log1pf(__expf(v)); }

// ---------------------------------------------------------------------------
// Pack W (M x K f32 row-major) into A-fragment tiles, f16:
//   layout [mt][kt][lane][16 halves], mt tiles M by 16, kt tiles K by 32.
// Branch-free: clamped loads + value select.
// ---------------------------------------------------------------------------
__global__ void pack_a_kernel(const float* __restrict__ W, _Float16* __restrict__ Wp,
                              int M, int K, int MT, int KT)
{
  int gid = blockIdx.x * blockDim.x + threadIdx.x;
  if (gid >= MT * KT * 32) return;
  int lane = gid & 31;
  int kt   = (gid >> 5) % KT;
  int mt   = gid / (KT * 32);
  int row  = mt * 16 + (lane & 15);
  int kb   = kt * 32 + ((lane < 16) ? 0 : 8);
  bool rv  = (row < M);
  const float* Wr = W + (size_t)(rv ? row : 0) * K;
  v16h tmp;
#pragma unroll
  for (int h = 0; h < 8; ++h) {
    int k1 = kb + h;
    int k2 = kb + 16 + h;
    float v1 = Wr[k1 < K ? k1 : 0];
    float v2 = Wr[k2 < K ? k2 : 0];
    tmp[h]     = (rv && k1 < K) ? (_Float16)v1 : (_Float16)0.f;
    tmp[h + 8] = (rv && k2 < K) ? (_Float16)v2 : (_Float16)0.f;
  }
  *((v16h*)Wp + gid) = tmp;
}

// ---------------------------------------------------------------------------
// Pack X (TOK_C x K f32, row stride ldx) into B-fragment tiles, f16:
//   layout [tt][kt][lane][16 halves]; col = lane&15, K in table order (8h+s, 16+8h+s-8).
// ---------------------------------------------------------------------------
__global__ void pack_b_kernel(const float* __restrict__ X, _Float16* __restrict__ Xp,
                              int K, int ldx, int KT)
{
  int gid = blockIdx.x * blockDim.x + threadIdx.x;
  if (gid >= TT_C * KT * 32) return;
  int lane = gid & 31;
  int kt   = (gid >> 5) % KT;
  int tt   = gid / (KT * 32);
  int tok  = tt * 16 + (lane & 15);
  int kb   = kt * 32 + ((lane < 16) ? 0 : 8);
  const float* Xr = X + (size_t)tok * ldx;
  v16h tmp;
#pragma unroll
  for (int h = 0; h < 16; ++h) {
    int k = kb + h + (h & 8);
    float v = Xr[k < K ? k : 0];
    tmp[h] = (k < K) ? (_Float16)v : (_Float16)0.f;
  }
  *((v16h*)Xp + gid) = tmp;
}

// ---------------------------------------------------------------------------
// GEMM on packed fragments: Y[t,m] = epi( sum_k X[t,k]*W[m,k] (+bias) )
// One wave -> one 16(M) x 16(T) tile; 4 waves/block along M share one B tile
// chain, staged into LDS by the Tensor Data Mover.
// EPI: 0 = linear(+bias), 2 = softplus(+bias), 3 = +resid
// ---------------------------------------------------------------------------
template<int EPI>
__global__ void wmma_gemm_kernel(const _Float16* __restrict__ Ap,
                                 const _Float16* __restrict__ Bp,
                                 const float* __restrict__ bias,
                                 const float* __restrict__ resid,
                                 float* __restrict__ Y,
                                 int M, int KT, int ldy)
{
  __shared__ v16h smemB[MAX_KT * 32];          // up to 16KB staged B tiles
  const int lane = threadIdx.x & 31;
  const int wid  = threadIdx.x >> 5;
  const int mt   = blockIdx.x * 4 + wid;
  const int tt   = blockIdx.y;

  const _Float16* bsrc = Bp + (size_t)tt * KT * 32 * 16;
#if USE_TDM
  if (wid == 0) {
    // 1-D TDM descriptor: KT*1024 bytes, 8-byte elements, contiguous.
    unsigned lds_addr = (unsigned)(uintptr_t)(&smemB[0]);   // low 32b = LDS offset
    unsigned long long ga = (unsigned long long)(uintptr_t)bsrc;
    unsigned n8 = (unsigned)(KT * 128);                     // elements of 8 bytes
    v4u g0;
    g0[0] = 1u;                                             // count=1 (valid), user mode
    g0[1] = lds_addr;                                       // lds_addr
    g0[2] = (unsigned)(ga & 0xFFFFFFFFu);                   // global_addr[31:0]
    g0[3] = (unsigned)((ga >> 32) & 0x1FFFFFFu) | (2u << 30); // global_addr[56:32], type=2
    v8i_ g1 = {};
    g1[0] = (int)(3u << 16);                                // data_size = 8B
    g1[1] = (int)((n8 & 0xFFFFu) << 16);                    // tensor_dim0[15:0]
    g1[2] = (int)(((n8 >> 16) & 0xFFFFu) | (1u << 16));     // tensor_dim0[31:16], tensor_dim1=1
    g1[3] = (int)((n8 & 0xFFFFu) << 16);                    // tile_dim0 = n8
    g1[5] = (int)n8;                                        // tensor_dim0_stride[31:0]
    v4i_ g2 = {};
    v4i_ g3 = {};
#if defined(__clang_major__) && __clang_major__ >= 23
    v8i_ g4 = {};
    __builtin_amdgcn_tensor_load_to_lds(g0, g1, g2, g3, g4, 0);
#else
    __builtin_amdgcn_tensor_load_to_lds(g0, g1, g2, g3, 0);
#endif
    __builtin_amdgcn_s_wait_tensorcnt(0);
  }
#else
  for (int i = threadIdx.x; i < KT * 32; i += (int)blockDim.x)
    smemB[i] = ((const v16h*)bsrc)[i];
#endif
  __syncthreads();                              // all waves reach this

  const int m0 = mt * 16;
  if (m0 >= M) return;                          // wave-uniform

  const v16h* a  = (const v16h*)Ap + (size_t)mt * KT * 32 + lane;
  const v16h* bl = smemB + lane;

  v8f acc = {};
#pragma unroll 4
  for (int kt = 0; kt < KT; ++kt) {
    v16h av = a[(size_t)kt * 32];               // 2x global_load_b128
    v16h bv = bl[kt * 32];                      // 2x ds_load_b128
    acc = __builtin_amdgcn_wmma_f32_16x16x32_f16(false, av, false, bv,
                                                 (short)0, acc, false, false);
  }

  const int t  = tt * 16 + (lane & 15);
  const int mb = m0 + ((lane < 16) ? 0 : 8);    // 8 contiguous output features
  float o[8];
#pragma unroll
  for (int r = 0; r < 8; ++r) o[r] = acc[r];

  if (m0 + 16 <= M) {                           // wave-uniform fast path
    if (bias) {
      float4 b0 = *(const float4*)(bias + mb);
      float4 b1 = *(const float4*)(bias + mb + 4);
      o[0] += b0.x; o[1] += b0.y; o[2] += b0.z; o[3] += b0.w;
      o[4] += b1.x; o[5] += b1.y; o[6] += b1.z; o[7] += b1.w;
    }
    if (EPI == 2) {
#pragma unroll
      for (int r = 0; r < 8; ++r) o[r] = softplus_f(o[r]);
    }
    float* yr = Y + (size_t)t * ldy + mb;
    if (EPI == 3) {
      float4 r0 = *(const float4*)(resid + (size_t)t * ldy + mb);
      float4 r1 = *(const float4*)(resid + (size_t)t * ldy + mb + 4);
      o[0] += r0.x; o[1] += r0.y; o[2] += r0.z; o[3] += r0.w;
      o[4] += r1.x; o[5] += r1.y; o[6] += r1.z; o[7] += r1.w;
    }
    *(float4*)(yr)     = make_float4(o[0], o[1], o[2], o[3]);
    *(float4*)(yr + 4) = make_float4(o[4], o[5], o[6], o[7]);
  } else {                                      // edge M tile
#pragma unroll
    for (int r = 0; r < 8; ++r) {
      int m = mb + r;
      if (m < M) {
        float v = o[r];
        if (bias) v += bias[m];
        if (EPI == 2) v = softplus_f(v);
        if (EPI == 3) v += resid[(size_t)t * ldy + m];
        Y[(size_t)t * ldy + m] = v;
      }
    }
  }
}

// ---------------------------------------------------------------------------
// RMSNorm: one wave per token, wave32 butterfly reduction.
// ---------------------------------------------------------------------------
__global__ void rmsnorm_kernel(const float* __restrict__ X, const float* __restrict__ w,
                               float* __restrict__ Y, int d)
{
  int tok  = (int)((blockIdx.x * blockDim.x + threadIdx.x) >> 5);
  int lane = threadIdx.x & 31;
  if (tok >= TOK_C) return;
  const float* xr = X + (size_t)tok * d;
  float ss = 0.f;
  for (int i = lane; i < d; i += 32) { float v = xr[i]; ss += v * v; }
#pragma unroll
  for (int off = 16; off > 0; off >>= 1) ss += __shfl_xor(ss, off, 32);
  float scale = rsqrtf(ss / (float)d + 1e-5f);
  float* yr = Y + (size_t)tok * d;
  for (int i = lane; i < d; i += 32) yr[i] = xr[i] * scale * w[i];
}

// ---------------------------------------------------------------------------
// Causal depthwise conv (width 4) + bias + SiLU. xz (T,2ED) cols 0..ED-1.
// ---------------------------------------------------------------------------
__global__ void dwconv_silu_kernel(const float* __restrict__ xz,
                                   const float* __restrict__ cw,
                                   const float* __restrict__ cb,
                                   float* __restrict__ xc,
                                   int L, int ED, int ld)
{
  int gid = blockIdx.x * blockDim.x + threadIdx.x;
  if (gid >= TOK_C * ED) return;
  int e = gid % ED;
  int t = gid / ED;          // b*L + l
  int l = t % L;
  float acc = cb[e];
#pragma unroll
  for (int j = 0; j < DCONV; ++j) {
    int ll = l - (DCONV - 1) + j;
    if (ll >= 0) acc += cw[e * DCONV + j] * xz[(size_t)(t - (DCONV - 1) + j) * ld + e];
  }
  xc[(size_t)t * ED + e] = silu_f(acc);
}

// ---------------------------------------------------------------------------
// Selective scan fused with D-skip and SiLU(z) gate; 16 states in VGPRs.
// ---------------------------------------------------------------------------
__global__ void scan_kernel(const float* __restrict__ xc,    // (T, ED)
                            const float* __restrict__ dl,    // (T, ED)
                            const float* __restrict__ dbc,   // (T, LD_DBC): [R..R+16)=B, [R+16..R+32)=C
                            const float* __restrict__ A_log, // (ED, 16)
                            const float* __restrict__ Dp,    // (ED)
                            const float* __restrict__ xz,    // (T, 2*ED); z at col ED+e
                            float* __restrict__ y,           // (T, ED)
                            int L, int ED, int R)
{
  int gid = blockIdx.x * blockDim.x + threadIdx.x;
  if (gid >= BATCH_C * ED) return;
  int e  = gid % ED;
  int bb = gid / ED;
  float A[NST], h[NST];
#pragma unroll
  for (int n = 0; n < NST; ++n) { A[n] = -__expf(A_log[e * NST + n]); h[n] = 0.f; }
  float Dd = Dp[e];
  const int ld2 = 2 * ED;
  size_t tb = (size_t)bb * L;
  for (int l = 0; l < L; ++l) {
    size_t t = tb + l;
    if (l + 16 < L) {   // gfx1250 global_prefetch_b8 hints down the L-chase
      __builtin_prefetch(dl  + (t + 16) * ED     + e, 0, 1);
      __builtin_prefetch(dbc + (t + 16) * LD_DBC + R, 0, 1);
    }
    float dt  = dl[t * ED + e];
    float xv  = xc[t * ED + e];
    float dbx = dt * xv;
    const float* bc = dbc + t * LD_DBC + R;
    float acc = 0.f;
#pragma unroll
    for (int n = 0; n < NST; ++n) {
      h[n] = __expf(dt * A[n]) * h[n] + dbx * bc[n];
      acc += h[n] * bc[NST + n];
    }
    acc += xv * Dd;
    float zv = xz[t * ld2 + ED + e];
    y[t * ED + e] = acc * silu_f(zv);
  }
}

// ---------------------------------------------------------------------------
// Layout shuffles + residual add
// ---------------------------------------------------------------------------
__global__ void ncl_to_nlc(const float* __restrict__ src, float* __restrict__ dst, int C, int L)
{
  int gid = blockIdx.x * blockDim.x + threadIdx.x;
  if (gid >= BATCH_C * C * L) return;
  int l = gid % L, c = (gid / L) % C, b = gid / (L * C);
  dst[((size_t)b * L + l) * C + c] = src[gid];
}

__global__ void nlc_to_ncl(const float* __restrict__ src, float* __restrict__ dst, int C, int L)
{
  int gid = blockIdx.x * blockDim.x + threadIdx.x;
  if (gid >= BATCH_C * C * L) return;
  int c = gid % C, l = (gid / C) % L, b = gid / (C * L);
  dst[((size_t)b * C + c) * L + l] = src[gid];
}

__global__ void add_inplace_kernel(float* __restrict__ a, const float* __restrict__ b, int n)
{
  int i = blockIdx.x * blockDim.x + threadIdx.x;
  if (i < n) a[i] += b[i];
}

// ---------------------------------------------------------------------------
// Host orchestration
// ---------------------------------------------------------------------------
struct LayerP {
  const float *norm_w, *in_w, *cw, *cb, *xp_w, *dt_w, *dt_b, *A_log, *D, *out_w;
};

extern "C" void kernel_launch(void* const* d_in, const int* in_sizes, int n_in,
                              void* d_out, int out_size, void* d_ws, size_t ws_size,
                              hipStream_t stream)
{
  (void)in_sizes; (void)n_in; (void)out_size; (void)ws_size;

  int idx = 0;
  auto nxt = [&]() { return (const float*)d_in[idx++]; };
  const float* x_in = nxt();
  auto nxtLayer = [&]() {
    LayerP p;
    p.norm_w = nxt(); p.in_w = nxt(); p.cw = nxt(); p.cb = nxt();
    p.xp_w = nxt();   p.dt_w = nxt(); p.dt_b = nxt(); p.A_log = nxt();
    p.D = nxt();      p.out_w = nxt();
    return p;
  };
  const float* down_cw[4]; const float* down_cb[4]; LayerP down_m[4][2];
  for (int i = 0; i < 4; ++i) {
    down_cw[i] = nxt(); down_cb[i] = nxt();
    down_m[i][0] = nxtLayer(); down_m[i][1] = nxtLayer();
  }
  LayerP mid_m[2][2];
  for (int j = 0; j < 2; ++j) { mid_m[j][0] = nxtLayer(); mid_m[j][1] = nxtLayer(); }
  const float* up_cw[4]; const float* up_cb[4]; LayerP up_m[4][2];
  for (int i = 0; i < 4; ++i) {
    up_cw[i] = nxt(); up_cb[i] = nxt();
    up_m[i][0] = nxtLayer(); up_m[i][1] = nxtLayer();
  }

  // ---- workspace carve-out ----
  float* wsf = (float*)d_ws;
  auto takef = [&](size_t n) { float* p = wsf; wsf += n; return p; };
  const size_t T = TOK_C;
  float* buf[2]; buf[0] = takef(T * 256); buf[1] = takef(T * 256);
  float* res[4]; for (int i = 0; i < 4; ++i) res[i] = takef(T * (size_t)(256 >> i));
  float* hS   = takef(T * 256);
  float* xzS  = takef(T * 1024);
  float* xcS  = takef(T * 512);
  float* dlS  = takef(T * 512);
  float* dbcS = takef(T * LD_DBC);
  float* yS   = takef(T * 512);
  _Float16* ApS = (_Float16*)takef(64 * 16 * 32 * 16 / 2 + 64);            // packed A
  _Float16* BpS = (_Float16*)takef((size_t)TT_C * MAX_KT * 32 * 16 / 2 + 64); // packed B

  // pack + WMMA GEMM
  auto gemm = [&](int epi, const float* X, const float* Wm, const float* bias,
                  const float* resid, float* Y, int M, int K, int ldx, int ldy) {
    int MT = (M + 15) / 16, KT = (K + 31) / 32;
    {
      int total = MT * KT * 32;
      pack_a_kernel<<<(total + 255) / 256, 256, 0, stream>>>(Wm, ApS, M, K, MT, KT);
    }
    {
      int total = TT_C * KT * 32;
      pack_b_kernel<<<(total + 255) / 256, 256, 0, stream>>>(X, BpS, K, ldx, KT);
    }
    dim3 g((unsigned)((MT + 3) / 4), (unsigned)TT_C, 1);
    dim3 b(128, 1, 1);
    switch (epi) {
      case 0: wmma_gemm_kernel<0><<<g, b, 0, stream>>>(ApS, BpS, bias, resid, Y, M, KT, ldy); break;
      case 2: wmma_gemm_kernel<2><<<g, b, 0, stream>>>(ApS, BpS, bias, resid, Y, M, KT, ldy); break;
      default: wmma_gemm_kernel<3><<<g, b, 0, stream>>>(ApS, BpS, bias, resid, Y, M, KT, ldy); break;
    }
  };

  auto run_block = [&](const float* xin, float* xout, int d, const LayerP& p) {
    const int ED = 2 * d;
    const int R  = (d + 15) / 16;
    const int RN = R + 2 * NST;
    { // RMSNorm -> hS
      int blocks = (int)((T * 32 + 255) / 256);
      rmsnorm_kernel<<<blocks, 256, 0, stream>>>(xin, p.norm_w, hS, d);
    }
    // in_proj: (T,d) x (2ED,d)^T -> xz (T, 2ED)
    gemm(0, hS, p.in_w, nullptr, nullptr, xzS, 2 * ED, d, d, 2 * ED);
    { // depthwise causal conv + SiLU -> xc (T, ED)
      int total = (int)T * ED;
      dwconv_silu_kernel<<<(total + 255) / 256, 256, 0, stream>>>(xzS, p.cw, p.cb, xcS,
                                                                  SEQ_C, ED, 2 * ED);
    }
    // x_proj -> dbc (T, RN), padded ld = LD_DBC
    gemm(0, xcS, p.xp_w, nullptr, nullptr, dbcS, RN, ED, ED, LD_DBC);
    // dt_proj + softplus -> delta (T, ED); reads first R cols of dbc
    gemm(2, dbcS, p.dt_w, p.dt_b, nullptr, dlS, ED, R, LD_DBC, ED);
    { // selective scan + D-skip + SiLU(z) gate -> y (T, ED)
      int total = BATCH_C * ED;
      scan_kernel<<<(total + 127) / 128, 128, 0, stream>>>(xcS, dlS, dbcS, p.A_log, p.D,
                                                           xzS, yS, SEQ_C, ED, R);
    }
    // out_proj + residual: xout = xin + y @ W^T
    gemm(3, yS, p.out_w, nullptr, xin, xout, d, ED, ED, d);
  };

  // ---- input (B,C,L) -> token-major (stored as residual level 0) ----
  {
    int total = BATCH_C * 256 * SEQ_C;
    ncl_to_nlc<<<(total + 255) / 256, 256, 0, stream>>>(x_in, res[0], 256, SEQ_C);
  }

  float* cur = res[0];
  int C = 256;
  auto nextbuf = [&](float* c) { return (c == buf[0]) ? buf[1] : buf[0]; };

  // ---- down path ----
  for (int i = 0; i < 4; ++i) {
    int co = C / 2;
    gemm(0, cur, down_cw[i], down_cb[i], nullptr, buf[0], co, C, C, co);
    run_block(buf[0], buf[1], co, down_m[i][0]);
    float* dst = (i < 3) ? res[i + 1] : buf[0];
    run_block(buf[1], dst, co, down_m[i][1]);
    cur = dst;
    C = co;
  }

  // ---- mid ----
  for (int j = 0; j < 2; ++j)
    for (int l = 0; l < 2; ++l) {
      float* nx = nextbuf(cur);
      run_block(cur, nx, C, mid_m[j][l]);
      cur = nx;
    }

  // ---- up path ----
  for (int i = 0; i < 4; ++i) {
    int co = C * 2;
    float* t1 = nextbuf(cur);
    gemm(0, cur, up_cw[i], up_cb[i], nullptr, t1, co, C, C, co);
    float* t2 = nextbuf(t1);
    run_block(t1, t2, co, up_m[i][0]);
    float* t3 = nextbuf(t2);
    run_block(t2, t3, co, up_m[i][1]);
    int n = (int)T * co;
    add_inplace_kernel<<<(n + 255) / 256, 256, 0, stream>>>(t3, res[3 - i], n);
    cur = t3;
    C = co;
  }

  // ---- token-major -> (B,C,L) output ----
  {
    int total = BATCH_C * 256 * SEQ_C;
    nlc_to_ncl<<<(total + 255) / 256, 256, 0, stream>>>(cur, (float*)d_out, 256, SEQ_C);
  }
}
